// DeepCLFM_42992622633202
// MI455X (gfx1250) — matrix operand order repaired and checked by reading, not compiled, not run on hardware
//
#include <hip/hip_runtime.h>
#include <hip/hip_bf16.h>

typedef __attribute__((ext_vector_type(16))) _Float16 v16h;
typedef __attribute__((ext_vector_type(8)))  float    v8f;

__device__ __forceinline__ v8f wmma16(v16h a, v16h b, v8f c) {
    return __builtin_amdgcn_wmma_f32_16x16x32_f16(false, a, false, b, (short)0, c, false, false);
}

__device__ __forceinline__ float sigmoidf_(float x) { return 1.f / (1.f + __expf(-x)); }
__device__ __forceinline__ float leaky_(float x) { return x > 0.f ? x : 0.01f * x; }

// ---------------------------------------------------------------------------
// Row-strip WMMA GEMM: C[M, NT*16] = A[M,K] @ W[NT*16, K]^T (+ bias)
// One wave owns 16 rows x full N. Each A fragment is loaded ONCE per k-step
// and reused across all NT column tiles (12x less A traffic than tile-per-wave).
// NT accumulators: NT*8 VGPRs (96 for NT=12) -- fine under wave32 budget.
// ---------------------------------------------------------------------------
template <int NT>
__global__ __launch_bounds__(32)
void gemm_rowstrip_kernel(const float* __restrict__ A, int lda,
                          const float* __restrict__ W, int ldw,
                          const float* __restrict__ bias,
                          float* __restrict__ C, int ldc,
                          int K)
{
    const int m0 = blockIdx.x * 16;
    const int lane = threadIdx.x;
    const int l15  = lane & 15;
    const int hgrp = lane >> 4;

    v8f acc[NT];
    #pragma unroll
    for (int nt = 0; nt < NT; ++nt) acc[nt] = (v8f){};

    const float* arow = A + (size_t)(m0 + l15) * lda;

    for (int k0 = 0; k0 < K; k0 += 32) {
        v16h a;
        #pragma unroll
        for (int i = 0; i < 8; ++i) {
            int kb = ((i < 4) ? 0 : 16) + hgrp * 8 + (i & 3) * 2;
            a[2*i]   = (_Float16)arow[k0 + kb];
            a[2*i+1] = (_Float16)arow[k0 + kb + 1];
        }
        #pragma unroll
        for (int nt = 0; nt < NT; ++nt) {
            const float* wrow = W + (size_t)(nt * 16 + l15) * ldw + k0 + 8 * hgrp;
            v16h b;
            #pragma unroll
            for (int e = 0; e < 16; ++e) b[e] = (_Float16)wrow[e < 8 ? e : e + 8];
            acc[nt] = wmma16(a, b, acc[nt]);
        }
    }
    #pragma unroll
    for (int nt = 0; nt < NT; ++nt) {
        float bv = bias ? bias[nt * 16 + l15] : 0.f;
        #pragma unroll
        for (int i = 0; i < 8; ++i) {
            int mm = m0 + i + 8 * hgrp;
            C[(size_t)mm * ldc + nt * 16 + l15] = acc[nt][i] + bv;
        }
    }
}

// ---------------------------------------------------------------------------
// GRU recurrence: one wave owns 16 batch rows, serial over T.
// Gi[b,t,0:192] precomputed (x@Wih.T + bih). Whh[192,64] staged to LDS f16.
// Per step: gh = h @ Whh.T via 12 WMMA N-tiles (K=64 -> 2 x K32), then gates.
// Prefetches next step's Gi tile (global_prefetch_b8) before the WMMA burst.
// ---------------------------------------------------------------------------
__global__ __launch_bounds__(32)
void gru_recur_kernel(const float* __restrict__ Gi,
                      const float* __restrict__ Whh,
                      const float* __restrict__ bhh,
                      float* __restrict__ Y, int ldy, int ycol,
                      float* __restrict__ Hout,
                      int T, int reverse)
{
    __shared__ _Float16 sW[192 * 64];   // Whh f16
    __shared__ _Float16 sH16[16 * 64];  // h as f16 (A-fragment source)
    __shared__ float    sH[16 * 64];    // h as f32
    __shared__ float    sGh[16 * 192];  // h @ Whh.T result
    __shared__ float    sb[192];        // bhh

    const int lane = threadIdx.x;
    const int b0 = blockIdx.x * 16;
    const int l15 = lane & 15;
    const int hgrp = lane >> 4;

    for (int idx = lane; idx < 192 * 64; idx += 32) sW[idx] = (_Float16)Whh[idx];
    for (int idx = lane; idx < 192; idx += 32) sb[idx] = bhh[idx];
    for (int idx = lane; idx < 16 * 64; idx += 32) { sH[idx] = 0.f; sH16[idx] = (_Float16)0.f; }
    __syncthreads();

    for (int step = 0; step < T; ++step) {
        const int t = reverse ? (T - 1 - step) : step;

        // Prefetch next step's Gi tile: 16 rows x 192 f32 = 96 cachelines(128B)
        if (step + 1 < T) {
            const int tn = reverse ? (T - 2 - step) : (step + 1);
            #pragma unroll
            for (int j = 0; j < 3; ++j) {
                int idx = j * 32 + lane;          // 0..95
                int row = idx / 6, line = idx % 6;
                const float* p = Gi + ((size_t)(b0 + row) * T + tn) * 192 + line * 32;
                __builtin_prefetch(p, 0, 3);      // -> global_prefetch_b8
            }
        }

        // A fragments (h tile, 16x64 -> two 16x32 fragments)
        v16h a0, a1;
        #pragma unroll
        for (int i = 0; i < 8; ++i) {
            int kb = ((i < 4) ? 0 : 16) + hgrp * 8 + (i & 3) * 2;
            a0[2*i]   = sH16[l15 * 64 + kb];
            a0[2*i+1] = sH16[l15 * 64 + kb + 1];
            a1[2*i]   = sH16[l15 * 64 + 32 + kb];
            a1[2*i+1] = sH16[l15 * 64 + 32 + kb + 1];
        }
        #pragma unroll
        for (int nt = 0; nt < 12; ++nt) {
            v16h bf0, bf1;
            const _Float16* wr = &sW[(nt * 16 + l15) * 64 + 8 * hgrp];
            #pragma unroll
            for (int e = 0; e < 16; ++e) { int ke = e < 8 ? e : e + 8; bf0[e] = wr[ke]; bf1[e] = wr[32 + ke]; }
            v8f c = {};
            c = wmma16(a0, bf0, c);
            c = wmma16(a1, bf1, c);
            #pragma unroll
            for (int i = 0; i < 8; ++i) {
                int mm = i + 8 * hgrp;
                sGh[mm * 192 + nt * 16 + l15] = c[i];
            }
        }
        __syncthreads();

        // Gate math over 16x64 elements (32 per lane, coalesced Gi reads)
        #pragma unroll
        for (int j = 0; j < 32; ++j) {
            int idx = j * 32 + lane;
            int row = idx >> 6, d = idx & 63;
            const float* gi = Gi + ((size_t)(b0 + row) * T + t) * 192;
            float ghr = sGh[row * 192 + d]        + sb[d];
            float ghz = sGh[row * 192 + 64 + d]   + sb[64 + d];
            float ghn = sGh[row * 192 + 128 + d]  + sb[128 + d];
            float r = sigmoidf_(gi[d] + ghr);
            float z = sigmoidf_(gi[64 + d] + ghz);
            float n = tanhf(gi[128 + d] + r * ghn);
            float hprev = sH[row * 64 + d];
            float hnew = (1.f - z) * n + z * hprev;
            sH[row * 64 + d] = hnew;
            sH16[row * 64 + d] = (_Float16)hnew;
            if (Y) Y[((size_t)(b0 + row) * T + t) * ldy + ycol + d] = hnew;
        }
        __syncthreads();
    }
    if (Hout) {
        for (int idx = lane; idx < 16 * 64; idx += 32) {
            int row = idx >> 6, d = idx & 63;
            Hout[(size_t)(b0 + row) * 64 + d] = sH[idx];
        }
    }
}

// ---------------------------------------------------------------------------
// Attention: w[b] = tanh(z[b]@A1.T + a1) @ A2.T + a2 ; beta = softmax_b(w);
// Out[b,:] = z[b,:] * beta[b].  One block of B threads (B = 512 or 1024).
// ---------------------------------------------------------------------------
__global__ void attn_kernel(const float* __restrict__ Z,
                            const float* __restrict__ A1, const float* __restrict__ a1,
                            const float* __restrict__ A2, const float* __restrict__ a2,
                            float* __restrict__ Out)
{
    __shared__ float red[1024];
    const int b = threadIdx.x;
    const int B = blockDim.x;
    const float* z = Z + (size_t)b * 64;

    float w = a2[0];
    #pragma unroll
    for (int j = 0; j < 8; ++j) {
        float acc = a1[j];
        for (int d = 0; d < 64; ++d) acc += z[d] * A1[j * 64 + d];
        w += tanhf(acc) * A2[j];
    }
    red[b] = w;
    __syncthreads();
    for (int s = B >> 1; s > 0; s >>= 1) {
        if (b < s) red[b] = fmaxf(red[b], red[b + s]);
        __syncthreads();
    }
    float mx = red[0];
    __syncthreads();
    float e = __expf(w - mx);
    red[b] = e;
    __syncthreads();
    for (int s = B >> 1; s > 0; s >>= 1) {
        if (b < s) red[b] += red[b + s];
        __syncthreads();
    }
    float beta = e / red[0];
    for (int d = 0; d < 64; ++d) Out[(size_t)b * 64 + d] = z[d] * beta;
}

// ---------------------------------------------------------------------------
// FM + embedding: x = [TEXT[b], LFM[b]] (128);
// lin = x@FW.T + Fb ; inter = 0.5*sum((x@V)^2 - (x^2)@(V^2));
// Emb[b] = [TEXT+LFM, sigmoid(lin+inter)].  One 64-thread block per b.
// ---------------------------------------------------------------------------
__global__ __launch_bounds__(64)
void fm_kernel(const float* __restrict__ TEXT, const float* __restrict__ LFM,
               const float* __restrict__ FW, const float* __restrict__ Fb,
               const float* __restrict__ FV, float* __restrict__ Emb)
{
    __shared__ float x[128];
    __shared__ float red[64];
    const int b = blockIdx.x, d = threadIdx.x;
    float tv = TEXT[(size_t)b * 64 + d], lv = LFM[(size_t)b * 64 + d];
    x[d] = tv; x[64 + d] = lv;
    __syncthreads();
    float lin = Fb[d], xv = 0.f, x2v2 = 0.f;
    for (int k = 0; k < 128; ++k) {
        float xk = x[k];
        lin += xk * FW[d * 128 + k];
        float v = FV[k * 64 + d];
        xv += xk * v;
        x2v2 += xk * xk * v * v;
    }
    red[d] = xv * xv - x2v2;
    __syncthreads();
    for (int s = 32; s > 0; s >>= 1) { if (d < s) red[d] += red[d + s]; __syncthreads(); }
    float inter = 0.5f * red[0];
    Emb[(size_t)b * 128 + d]      = tv + lv;
    Emb[(size_t)b * 128 + 64 + d] = sigmoidf_(lin + inter);
}

// ---------------------------------------------------------------------------
// Pair scoring: x1 = leaky(PU[u]+PI[i]) (b1 folded into PU);
// x2 = leaky(x1 @ W2.T + b2) via WMMA (16 pairs x 16 dims, K=64);
// score = leaky(x2 @ Wo.T + bo).  8 waves/block, 16 pairs/wave.
// ---------------------------------------------------------------------------
__global__ __launch_bounds__(256)
void pair_kernel(const float* __restrict__ PU, const float* __restrict__ PI,
                 const float* __restrict__ W2, const float* __restrict__ b2,
                 const float* __restrict__ Wo, const float* __restrict__ bo,
                 float* __restrict__ Out, int nI)
{
    __shared__ _Float16 sx[8][16 * 64];
    const int wave = threadIdx.x >> 5, lane = threadIdx.x & 31;
    const long p0 = ((long)blockIdx.x * 8 + wave) * 16;  // nI multiple of 16
    const int u = (int)(p0 / nI);
    const int i0 = (int)(p0 % nI);
    _Float16* X = sx[wave];
    const float* pu = PU + (size_t)u * 64;

    #pragma unroll
    for (int j = 0; j < 32; ++j) {
        int idx = j * 32 + lane;
        int row = idx >> 6, d = idx & 63;
        float v = pu[d] + PI[(size_t)(i0 + row) * 64 + d];
        X[idx] = (_Float16)leaky_(v);
    }
    __syncthreads();

    const int l15 = lane & 15, hgrp = lane >> 4;
    v16h a0, a1, bf0, bf1;
    #pragma unroll
    for (int i = 0; i < 8; ++i) {
        int kb = ((i < 4) ? 0 : 16) + hgrp * 8 + (i & 3) * 2;
        a0[2*i]   = X[l15 * 64 + kb];
        a0[2*i+1] = X[l15 * 64 + kb + 1];
        a1[2*i]   = X[l15 * 64 + 32 + kb];
        a1[2*i+1] = X[l15 * 64 + 32 + kb + 1];
    }
    const float* wr = W2 + l15 * 64 + 8 * hgrp;
    #pragma unroll
    for (int e = 0; e < 16; ++e) { int ke = e < 8 ? e : e + 8; bf0[e] = (_Float16)wr[ke]; bf1[e] = (_Float16)wr[32 + ke]; }

    v8f c = {};
    c = wmma16(a0, bf0, c);
    c = wmma16(a1, bf1, c);

    const float b2v = b2[l15], wov = Wo[l15], bov = bo[0];
    #pragma unroll
    for (int i = 0; i < 8; ++i) {
        float v = leaky_(c[i] + b2v) * wov;
        v += __shfl_xor(v, 1, 16);
        v += __shfl_xor(v, 2, 16);
        v += __shfl_xor(v, 4, 16);
        v += __shfl_xor(v, 8, 16);
        float s = leaky_(v + bov);
        if (l15 == 0) Out[p0 + i + 8 * hgrp] = s;
    }
}

// ---------------------------------------------------------------------------
// Host orchestration
// ---------------------------------------------------------------------------
static inline void run_gemm192(const float* A, int lda, const float* W, int ldw,
                               const float* bias, float* C, int M, int K, hipStream_t s)
{
    gemm_rowstrip_kernel<12><<<dim3(M / 16), dim3(32), 0, s>>>(A, lda, W, ldw, bias, C, 192, K);
}
static inline void run_gemm64(const float* A, int lda, const float* W, int ldw,
                              const float* bias, float* C, int M, int K, hipStream_t s)
{
    gemm_rowstrip_kernel<4><<<dim3(M / 16), dim3(32), 0, s>>>(A, lda, W, ldw, bias, C, 64, K);
}

extern "C" void kernel_launch(void* const* d_in, const int* in_sizes, int n_in,
                              void* d_out, int out_size, void* d_ws, size_t ws_size,
                              hipStream_t stream)
{
    (void)in_sizes; (void)n_in; (void)out_size; (void)ws_size;
    const int D = 64, UC = 512, IC = 1024, T = 100;

    const float* user_words = (const float*)d_in[0];
    const float* item_words = (const float*)d_in[1];
    const float* LFM_U = (const float*)d_in[2];
    const float* LFM_I = (const float*)d_in[3];
    const float* gruw[2][8]; // [u,i][W0,U0,b0,c0,W1,U1,b1,c1]
    for (int p = 0; p < 2; ++p)
        for (int k = 0; k < 8; ++k) gruw[p][k] = (const float*)d_in[4 + p * 8 + k];
    const float* attw[2][7]; // [u,i][A1,a1,A2,a2,FW,Fb,FV]
    for (int p = 0; p < 2; ++p)
        for (int k = 0; k < 7; ++k) attw[p][k] = (const float*)d_in[20 + p * 7 + k];
    const float* W1 = (const float*)d_in[34];
    const float* b1 = (const float*)d_in[35];
    const float* W2 = (const float*)d_in[36];
    const float* b2 = (const float*)d_in[37];
    const float* Wo = (const float*)d_in[38];
    const float* bo = (const float*)d_in[39];

    // Workspace layout (sized for the larger entity, IC=1024)
    char* ws = (char*)d_ws;
    size_t o = 0;
    float* Gi   = (float*)(ws + o); o += (size_t)IC * T * 192 * 4;   // 78.6 MB (reused)
    float* Yb   = (float*)(ws + o); o += (size_t)IC * T * 128 * 4;   // 52.4 MB (reused)
    float* hb   = (float*)(ws + o); o += (size_t)IC * 64 * 4;
    float* TEXT = (float*)(ws + o); o += (size_t)IC * 64 * 4;
    float* embU = (float*)(ws + o); o += (size_t)UC * 128 * 4;
    float* embI = (float*)(ws + o); o += (size_t)IC * 128 * 4;
    float* PU   = (float*)(ws + o); o += (size_t)UC * 64 * 4;
    float* PI   = (float*)(ws + o); o += (size_t)IC * 64 * 4;

    const float* Xw[2]  = { user_words, item_words };
    const float* LFM[2] = { LFM_U, LFM_I };
    float* EMB[2] = { embU, embI };
    const int Bsz[2] = { UC, IC };

    for (int p = 0; p < 2; ++p) {
        const int B = Bsz[p], MT = B * T, nblk = B / 16;
        const float *W0 = gruw[p][0], *U0 = gruw[p][1], *b0c = gruw[p][2], *c0 = gruw[p][3];
        const float *Wl1 = gruw[p][4], *Ul1 = gruw[p][5], *bl1 = gruw[p][6], *cl1 = gruw[p][7];

        // layer 0 forward: input proj + recurrence (writes Y cols 0..63)
        run_gemm192(Xw[p], D, W0, D, b0c, Gi, MT, D, stream);
        gru_recur_kernel<<<nblk, 32, 0, stream>>>(Gi, U0, c0, Yb, 128, 0, nullptr, T, 0);
        // layer 0 backward (writes Y cols 64..127)
        run_gemm192(Xw[p], D, W0 + 192 * D, D, b0c + 192, Gi, MT, D, stream);
        gru_recur_kernel<<<nblk, 32, 0, stream>>>(Gi, U0 + 192 * D, c0 + 192, Yb, 128, 64,
                                                  nullptr, T, 1);
        // layer 1 backward only (forward direction is dead code in the reference)
        run_gemm192(Yb, 128, Wl1 + 192 * 128, 128, bl1 + 192, Gi, MT, 128, stream);
        gru_recur_kernel<<<nblk, 32, 0, stream>>>(Gi, Ul1 + 192 * D, cl1 + 192,
                                                  nullptr, 0, 0, hb, T, 1);
        // attention scale (softmax over batch), then FM -> embedding [B,128]
        attn_kernel<<<1, B, 0, stream>>>(hb, attw[p][0], attw[p][1], attw[p][2], attw[p][3], TEXT);
        fm_kernel<<<B, 64, 0, stream>>>(TEXT, LFM[p], attw[p][4], attw[p][5], attw[p][6], EMB[p]);
    }

    // Pair MLP layer 1 is separable: PU = U_emb@W1[:, :128].T + b1 ; PI = I_emb@W1[:,128:].T
    run_gemm64(embU, 128, W1, 256, b1, PU, UC, 128, stream);
    run_gemm64(embI, 128, W1 + 128, 256, nullptr, PI, IC, 128, stream);

    // Pair scoring: 524288 pairs, 128 per block
    const int npairs = UC * IC;
    pair_kernel<<<npairs / 128, 256, 0, stream>>>(PU, PI, W2, b2, Wo, bo, (float*)d_out, IC);
}
